// denoising_model_24764781429264
// MI455X (gfx1250) — matrix operand order repaired and checked by reading, not compiled, not run on hardware
//
#include <hip/hip_runtime.h>
#include <cstdint>
#include <cstddef>

#define N_NODES 50000
#define NFEAT 256
#define NLABEL 32
#define NHID 256
#define GDEPTH 4
#define FDIM 288   // NHID + NLABEL

typedef __attribute__((ext_vector_type(16))) _Float16 v16h;
typedef __attribute__((ext_vector_type(8)))  float    v8f;

__device__ __forceinline__ void atomAddF(float* p, float v) {
    // relaxed, device scope -> global_atomic_add_f32 (no CAS loop)
    __hip_atomic_fetch_add(p, v, __ATOMIC_RELAXED, __HIP_MEMORY_SCOPE_AGENT);
}

// ---------------------------------------------------------------------------
// small utility kernels
// ---------------------------------------------------------------------------

// W [K, Nout] f32  ->  WT [Nout, K] f16   (weights are tiny; done once per call)
__global__ void k_convert_transpose(const float* __restrict__ W,
                                    _Float16* __restrict__ WT, int K, int Nout) {
    int idx = blockIdx.x * blockDim.x + threadIdx.x;
    if (idx >= K * Nout) return;
    int k = idx / Nout, n = idx - k * Nout;
    WT[(size_t)n * K + k] = (_Float16)W[idx];
}

__global__ void k_deg_init(float* __restrict__ deg, int n) {
    int i = blockIdx.x * blockDim.x + threadIdx.x;
    if (i < n) deg[i] = 2.0f;                    // improved GCN: +2 self-loop weight
}

__global__ void k_deg_count(const int* __restrict__ dst, float* __restrict__ deg, int E) {
    int e = blockIdx.x * blockDim.x + threadIdx.x;
    if (e < E) atomAddF(&deg[dst[e]], 1.0f);
}

__global__ void k_dinv(const float* __restrict__ deg, float* __restrict__ dinv, int n) {
    int i = blockIdx.x * blockDim.x + threadIdx.x;
    if (i < n) dinv[i] = rsqrtf(deg[i]);
}

// per-edge normalization coefficient, computed once and reused by all 4 layers
__global__ void k_coef(const int* __restrict__ src, const int* __restrict__ dst,
                       const float* __restrict__ dinv, float* __restrict__ coef, int E) {
    int e = blockIdx.x * blockDim.x + threadIdx.x;
    if (e < E) coef[e] = dinv[src[e]] * dinv[dst[e]];
}

// sinusoidal positional embedding -> f16 [N, 256] : [sin(128) | cos(128)]
__global__ void k_emb(const float* __restrict__ t, _Float16* __restrict__ emb, int n) {
    int i = blockIdx.x * blockDim.x + threadIdx.x;
    int node = i >> 7, j = i & 127;
    if (node >= n) return;
    float x = t[node] * 4.0f;                            // t/steps*steps*RESCALE == t*4
    float freq = expf(-0.07252236513367073f * (float)j); // -ln(10000)/127
    float ang = x * freq;
    emb[(size_t)node * 256 + j]       = (_Float16)sinf(ang);
    emb[(size_t)node * 256 + 128 + j] = (_Float16)cosf(ang);
}

// h16 = concat(x, qY) as f16 [N, 288]; label columns written once (never change)
__global__ void k_concat(const float* __restrict__ x, const float* __restrict__ q,
                         _Float16* __restrict__ h16, int n) {
    int i = blockIdx.x * blockDim.x + threadIdx.x;
    if (i >= n * FDIM) return;
    int node = i / FDIM, c = i - node * FDIM;
    float v = (c < NFEAT) ? x[(size_t)node * NFEAT + c]
                          : q[(size_t)node * NLABEL + (c - NFEAT)];
    h16[i] = (_Float16)v;
}

// one wave per edge; lane-coalesced float atomics: agg[dst] += coef[e] * xw[src]
__global__ void k_scatter(const int* __restrict__ src, const int* __restrict__ dst,
                          const float* __restrict__ coef,
                          const float* __restrict__ xw,
                          float* __restrict__ agg, int E) {
    int e = blockIdx.x * (blockDim.x >> 5) + (threadIdx.x >> 5);
    if (e >= E) return;
    int lane = threadIdx.x & 31;
    int s = src[e], d = dst[e];
    float c = coef[e];
    const float* xs = xw  + (size_t)s * NHID;
    float*       ad = agg + (size_t)d * NHID;
#pragma unroll
    for (int j = 0; j < NHID / 32; ++j) {
        int f = lane + j * 32;
        atomAddF(&ad[f], xs[f] * c);
    }
}

// h16[:, :256] = relu(agg + 2*dinv^2*xw + gcn_b + skipout)   (skipout already has skip_b + temb)
__global__ void k_combine(const float* __restrict__ agg, const float* __restrict__ xw,
                          const float* __restrict__ dinv, const float* __restrict__ gcn_b,
                          const float* __restrict__ skipout, _Float16* __restrict__ h16) {
    int node = blockIdx.x, f = threadIdx.x;
    size_t idx = (size_t)node * NHID + f;
    float s = dinv[node];
    float pre = agg[idx] + 2.0f * s * s * xw[idx] + gcn_b[f] + skipout[idx];
    h16[(size_t)node * FDIM + f] = (_Float16)fmaxf(pre, 0.0f);
}

// ---------------------------------------------------------------------------
// WMMA GEMM:  out[M, Nout] = act( A[M,K](f16) * W(f16, pre-transposed [Nout,K]) + bias + addend )
// One wave computes a 16 x (16*NT) strip of C, accumulating in f32 via
// v_wmma_f32_16x16x32_f16. 4 waves / block over M. M must be a multiple of 16
// (N_NODES = 50000 = 3125*16). Fragment layouts per CDNA5 ISA 7.12.2.
// Ping-pong double buffering (unroll-by-2, two disjoint register sets): loads
// for stage s+1 issue before the WMMAs of stage s, no stage-rotate moves, no
// WMMA->VALU WAR-hazard NOPs from register copies.
// ---------------------------------------------------------------------------
template <int NT>
__global__ void k_gemm(const _Float16* __restrict__ A, int lda,
                       const _Float16* __restrict__ BT, int K,
                       const float* __restrict__ bias,
                       const float* __restrict__ addend, int addld,
                       float* __restrict__ outf, int outld,
                       _Float16* __restrict__ outh, int outhld,
                       int M, int act) {
    const int wave = threadIdx.x >> 5;
    const int lane = threadIdx.x & 31;
    const int m0 = (blockIdx.x * (blockDim.x >> 5) + wave) * 16;
    if (m0 >= M) return;                        // wave-uniform: EXEC stays all-ones
    const int n0   = blockIdx.y * (16 * NT);
    const int half = lane >> 4;                 // which 16-lane half of the wave
    const int mr   = lane & 15;                 // A row / B column within tile

    v8f acc[NT];
#pragma unroll
    for (int t = 0; t < NT; ++t)
#pragma unroll
        for (int r = 0; r < 8; ++r) acc[t][r] = 0.0f;

    // A 16x32 f16 fragment: lanes 0-15 hold K {0..7, 16..23}; lanes 16-31 hold {8..15, 24..31}
    const _Float16* arow = A + (size_t)(m0 + mr) * lda + half * 8;
    // B 32x16 f16 fragment, column-major per lane: lanes 0-15 K {0..7, 16..23}; lanes 16-31 K {8..15, 24..31}
    const _Float16* brow = BT + (size_t)(n0 + mr) * K + half * 8;

    union Fr { v16h h; uint4 u[2]; };
    Fr a[2], b[2][NT];

    auto loadStage = [&](int st, int kk) {
        a[st].u[0] = *reinterpret_cast<const uint4*>(arow + kk);
        a[st].u[1] = *reinterpret_cast<const uint4*>(arow + kk + 16);
#pragma unroll
        for (int t = 0; t < NT; ++t) {
            const _Float16* bp = brow + (size_t)t * 16 * K + kk;
            b[st][t].u[0] = *reinterpret_cast<const uint4*>(bp);
            b[st][t].u[1] = *reinterpret_cast<const uint4*>(bp + 16);
        }
    };
    auto mmaStage = [&](int st) {
#pragma unroll
        for (int t = 0; t < NT; ++t)
            acc[t] = __builtin_amdgcn_wmma_f32_16x16x32_f16(
                false, a[st].h, false, b[st][t].h, (short)0, acc[t], false, false);
    };

    const int ks = K >> 5;                      // k-steps (8 / 9 / 16 / 18 here)
    loadStage(0, 0);
    int s = 0;
    for (; s + 2 <= ks; s += 2) {
        loadStage(1, (s + 1) << 5);             // prefetch odd stage
        mmaStage(0);                            // consume even stage
        if (s + 2 < ks) loadStage(0, (s + 2) << 5);
        mmaStage(1);
    }
    if (s < ks) mmaStage(0);                    // odd-count tail (K=288 -> 9 steps)

    // epilogue; C/D layout: VGPR r -> M = m0 + r + 8*half, N = n0 + t*16 + (lane&15)
#pragma unroll
    for (int t = 0; t < NT; ++t) {
        int n = n0 + t * 16 + mr;
        float bv = bias ? bias[n] : 0.0f;
#pragma unroll
        for (int r = 0; r < 8; ++r) {
            int m = m0 + r + 8 * half;
            float v = acc[t][r] + bv;
            if (addend) v += addend[(size_t)m * addld + n];
            if (act == 1) v = (v > 0.0f) ? v : expm1f(v);       // ELU
            else if (act == 2) v = fmaxf(v, 0.0f);              // ReLU
            if (outf) outf[(size_t)m * outld + n] = v;
            if (outh) outh[(size_t)m * outhld + n] = (_Float16)v;
        }
    }
}

// ---------------------------------------------------------------------------

extern "C" void kernel_launch(void* const* d_in, const int* in_sizes, int n_in,
                              void* d_out, int out_size, void* d_ws, size_t ws_size,
                              hipStream_t stream) {
    const float* x      = (const float*)d_in[0];
    const float* qY     = (const float*)d_in[1];
    const int*   adj    = (const int*)  d_in[2];   // [2, E]
    const float* t      = (const float*)d_in[3];
    const float* gcn_W  = (const float*)d_in[5];
    const float* gcn_b  = (const float*)d_in[6];
    const float* skip_W = (const float*)d_in[7];
    const float* skip_b = (const float*)d_in[8];
    const float* tm_W1  = (const float*)d_in[9];
    const float* tm_b1  = (const float*)d_in[10];
    const float* tm_W2  = (const float*)d_in[11];
    const float* tm_b2  = (const float*)d_in[12];
    const float* fin_W1 = (const float*)d_in[13];
    const float* fin_b1 = (const float*)d_in[14];
    const float* fin_W2 = (const float*)d_in[15];
    const float* fin_b2 = (const float*)d_in[16];
    float* out = (float*)d_out;

    const int E = in_sizes[2] / 2;
    const int N = in_sizes[3];

    // ---- workspace carve (256B aligned); ~240 MB total with aliasing ----
    uint8_t* ws = (uint8_t*)d_ws;
    size_t off = 0;
    auto take = [&](size_t bytes) -> void* {
        void* p = ws + off;
        off = (off + bytes + 255) & ~(size_t)255;
        return p;
    };
    _Float16* gcnT  = (_Float16*)take((size_t)GDEPTH * FDIM * NHID * 2);
    _Float16* skipT = (_Float16*)take((size_t)GDEPTH * FDIM * NHID * 2);
    _Float16* tm1T  = (_Float16*)take((size_t)NHID * 2 * NHID * 2);
    _Float16* tm2T  = (_Float16*)take((size_t)2 * NHID * NHID * 2);
    _Float16* fin1T = (_Float16*)take((size_t)FDIM * 2 * FDIM * 2);
    _Float16* fin2T = (_Float16*)take((size_t)2 * FDIM * NLABEL * 2);
    float* deg   = (float*)take((size_t)N * 4);
    float* dinv  = (float*)take((size_t)N * 4);
    float* coefE = (float*)take((size_t)E * 4);
    float* temb  = (float*)take((size_t)N * NHID * 4);
    _Float16* h16 = (_Float16*)take((size_t)N * FDIM * 2);
    float* bufA = (float*)take((size_t)N * NHID * 4);   // xw   | alias: emb16, f1_16 head
    float* bufB = (float*)take((size_t)N * NHID * 4);   // skip | alias: t1_16, f1_16 tail
    float* bufC = (float*)take((size_t)N * NHID * 4);   // agg
    _Float16* emb16 = (_Float16*)bufA;                  // N*256*2  <= bufA
    _Float16* t116  = (_Float16*)bufB;                  // N*512*2  == bufB
    _Float16* f116  = (_Float16*)bufA;                  // N*576*2 spans bufA+bufB (contiguous)

    // ---- weight conversion / transpose to f16 [Nout, K] ----
    auto cw = [&](const float* W, _Float16* WT, int K, int Nout) {
        int total = K * Nout;
        k_convert_transpose<<<(total + 255) / 256, 256, 0, stream>>>(W, WT, K, Nout);
    };
    for (int i = 0; i < GDEPTH; ++i) {
        cw(gcn_W  + (size_t)i * FDIM * NHID, gcnT  + (size_t)i * FDIM * NHID, FDIM, NHID);
        cw(skip_W + (size_t)i * FDIM * NHID, skipT + (size_t)i * FDIM * NHID, FDIM, NHID);
    }
    cw(tm_W1,  tm1T,  NHID, 2 * NHID);
    cw(tm_W2,  tm2T,  2 * NHID, NHID);
    cw(fin_W1, fin1T, FDIM, 2 * FDIM);
    cw(fin_W2, fin2T, 2 * FDIM, NLABEL);

    // ---- degree / normalization / per-edge coef (graph is static: do once) ----
    k_deg_init <<<(N + 255) / 256, 256, 0, stream>>>(deg, N);
    k_deg_count<<<(E + 255) / 256, 256, 0, stream>>>(adj + E, deg, E);   // dst row
    k_dinv     <<<(N + 255) / 256, 256, 0, stream>>>(deg, dinv, N);
    k_coef     <<<(E + 255) / 256, 256, 0, stream>>>(adj, adj + E, dinv, coefE, E);

    const int rowBlocks = (N / 16 + 3) / 4;   // 4 waves per 128-thread block

    auto gemm4 = [&](const _Float16* A, int lda, const _Float16* BT, int K, int Nout,
                     const float* bias, const float* addend, int addld,
                     float* outf, int outld, _Float16* outh, int outhld, int act) {
        dim3 grid(rowBlocks, Nout / 64);
        k_gemm<4><<<grid, 128, 0, stream>>>(A, lda, BT, K, bias, addend, addld,
                                            outf, outld, outh, outhld, N, act);
    };

    // ---- time embedding MLP: emb -> 512 (ELU) -> 256 ----
    k_emb<<<((size_t)N * 128 + 255) / 256, 256, 0, stream>>>(t, emb16, N);
    gemm4(emb16, NHID, tm1T, NHID, 2 * NHID, tm_b1, nullptr, 0,
          nullptr, 0, t116, 2 * NHID, /*ELU*/1);
    gemm4(t116, 2 * NHID, tm2T, 2 * NHID, NHID, tm_b2, nullptr, 0,
          temb, NHID, nullptr, 0, /*none*/0);

    // ---- h0 = concat(x, qY) ----
    k_concat<<<((size_t)N * FDIM + 255) / 256, 256, 0, stream>>>(x, qY, h16, N);

    // ---- GCN layers ----
    for (int i = 0; i < GDEPTH; ++i) {
        // xw = h @ gcn_W[i]
        gemm4(h16, FDIM, gcnT + (size_t)i * FDIM * NHID, FDIM, NHID,
              nullptr, nullptr, 0, bufA, NHID, nullptr, 0, 0);
        // skipout = h @ skip_W[i] + skip_b[i] + temb
        gemm4(h16, FDIM, skipT + (size_t)i * FDIM * NHID, FDIM, NHID,
              skip_b + (size_t)i * NHID, temb, NHID, bufB, NHID, nullptr, 0, 0);
        // agg = segment_sum(coef * xw[src] -> dst)
        hipMemsetAsync(bufC, 0, (size_t)N * NHID * 4, stream);
        k_scatter<<<(E + 7) / 8, 256, 0, stream>>>(adj, adj + E, coefE, bufA, bufC, E);
        // h = relu(agg + 2*dinv^2*xw + gcn_b + skipout); write f16 cols 0..255
        k_combine<<<N, NHID, 0, stream>>>(bufC, bufA, dinv,
                                          gcn_b + (size_t)i * NHID, bufB, h16);
    }

    // ---- final MLP: 288 -> 576 (ELU) -> 32 ----
    gemm4(h16, FDIM, fin1T, FDIM, 2 * FDIM, fin_b1, nullptr, 0,
          nullptr, 0, f116, 2 * FDIM, /*ELU*/1);
    dim3 g2(rowBlocks, 1);
    k_gemm<2><<<g2, 128, 0, stream>>>(f116, 2 * FDIM, fin2T, 2 * FDIM, fin_b2,
                                      nullptr, 0, out, NLABEL, nullptr, 0, N, 0);

    (void)n_in; (void)ws_size; (void)out_size;
}
